// Decoder_5927054868856
// MI455X (gfx1250) — matrix operand order repaired and checked by reading, not compiled, not run on hardware
//
#include <hip/hip_runtime.h>

// ---------------------------------------------------------------------------
// Decoder: per-step Bahdanau attention + LSTM + vocab projection (WMMA bf16)
// B=32, T=64, S=128, D=256, V=32000
// ---------------------------------------------------------------------------

#define B_  32
#define T_  64
#define S_  128
#define D_  256
#define V_  32000

typedef __bf16 bf16x16 __attribute__((ext_vector_type(16)));
typedef float  f32x8   __attribute__((ext_vector_type(8)));

__device__ __forceinline__ __bf16 f2bf(float f) {
    union { float f; unsigned u; } a; a.f = f;
    unsigned r = a.u + 0x7FFFu + ((a.u >> 16) & 1u);   // round-to-nearest-even
    union { unsigned short s; __bf16 b; } o;
    o.s = (unsigned short)(r >> 16);
    return o.b;
}

__device__ __forceinline__ float sigm(float x) { return 1.0f / (1.0f + expf(-x)); }

// ---------------------------------------------------------------------------
// Setup: transpose Wo [D][V] fp32 -> WoT [V][D] bf16 (L2-resident GEMM operand)
// grid (V/32, D/32), block 256 (= 32 x 8)
// ---------------------------------------------------------------------------
__global__ void __launch_bounds__(256)
k_transpose_wo(const float* __restrict__ Wo, __bf16* __restrict__ WoT) {
    __shared__ float tile[32][33];
    const int tn = blockIdx.x;           // V tile (0..999)
    const int tk = blockIdx.y;           // D tile (0..7)
    const int lx = threadIdx.x & 31;
    const int ly = threadIdx.x >> 5;     // 0..7
    #pragma unroll
    for (int r = 0; r < 32; r += 8)
        tile[ly + r][lx] = Wo[(size_t)(tk * 32 + ly + r) * V_ + tn * 32 + lx];
    __syncthreads();
    #pragma unroll
    for (int r = 0; r < 32; r += 8)
        WoT[(size_t)(tn * 32 + ly + r) * D_ + tk * 32 + lx] = f2bf(tile[lx][ly + r]);
}

// ---------------------------------------------------------------------------
// Setup: keys = enc_out @ Wk + bk       [B*S][D], one block per (b,s) row
// ---------------------------------------------------------------------------
__global__ void __launch_bounds__(256)
k_keys(const float* __restrict__ enc_out, const float* __restrict__ Wk,
       const float* __restrict__ bk, float* __restrict__ keys) {
    const int bs = blockIdx.x;           // 0 .. B*S-1
    const int tid = threadIdx.x;         // d
    __shared__ float esh[D_];
    esh[tid] = enc_out[(size_t)bs * D_ + tid];
    __syncthreads();
    float acc = bk[tid];
    for (int k = 0; k < D_; ++k) acc += esh[k] * Wk[k * D_ + tid];
    keys[(size_t)bs * D_ + tid] = acc;
}

// ---------------------------------------------------------------------------
// Setup: copy initial LSTM state into workspace
// ---------------------------------------------------------------------------
__global__ void __launch_bounds__(256)
k_init_state(const float* __restrict__ h0, const float* __restrict__ c0,
             float* __restrict__ h, float* __restrict__ c) {
    const int i = blockIdx.x * 256 + threadIdx.x;   // grid = 32 -> 8192
    h[i] = h0[i];
    c[i] = c0[i];
}

// ---------------------------------------------------------------------------
// Per step: attention + LSTM cell, fused. One block per batch row.
// Writes h (fp32), c (fp32), hbf (bf16 copy of h_new for the WMMA GEMM).
// ---------------------------------------------------------------------------
__global__ void __launch_bounds__(256)
k_attn_lstm(const int* __restrict__ tokens, int t,
            float* __restrict__ h, float* __restrict__ c,
            const float* __restrict__ keys, const float* __restrict__ enc_out,
            const float* __restrict__ E,
            const float* __restrict__ Wq, const float* __restrict__ bq,
            const float* __restrict__ Vw, const float* __restrict__ bv,
            const float* __restrict__ Wx, const float* __restrict__ Wh,
            const float* __restrict__ b_lstm,
            __bf16* __restrict__ hbf) {
    const int b = blockIdx.x;
    const int tid = threadIdx.x;
    __shared__ float hsh[D_], qsh[D_], xsh[2 * D_], sc[S_], red[256];

    hsh[tid] = h[b * D_ + tid];
    __syncthreads();

    // q = h @ Wq + bq
    {
        float acc = bq[tid];
        for (int k = 0; k < D_; ++k) acc += hsh[k] * Wq[k * D_ + tid];
        qsh[tid] = acc;
    }
    __syncthreads();

    // score[s] = sum_d tanh(q[d] + keys[b,s,d]) * Vw[d] + bv
    if (tid < S_) {
        const float* kr = keys + ((size_t)b * S_ + tid) * D_;
        float s = bv[0];
        for (int d = 0; d < D_; ++d) s += tanhf(qsh[d] + kr[d]) * Vw[d];
        sc[tid] = s;
    }
    __syncthreads();

    // softmax over S
    float v = (tid < S_) ? sc[tid] : -3.0e38f;
    red[tid] = v; __syncthreads();
    for (int s = 128; s > 0; s >>= 1) { if (tid < s) red[tid] = fmaxf(red[tid], red[tid + s]); __syncthreads(); }
    const float mx = red[0]; __syncthreads();
    float e = (tid < S_) ? expf(sc[tid] - mx) : 0.0f;
    red[tid] = e; __syncthreads();
    for (int s = 128; s > 0; s >>= 1) { if (tid < s) red[tid] += red[tid + s]; __syncthreads(); }
    const float inv = 1.0f / red[0];
    __syncthreads();
    if (tid < S_) sc[tid] = e * inv;
    __syncthreads();

    // ctx[d] = sum_s attn[s] * enc_out[b,s,d];  x = [ctx, emb]
    {
        float ctx = 0.0f;
        const float* er = enc_out + (size_t)b * S_ * D_ + tid;
        for (int s = 0; s < S_; ++s) ctx += sc[s] * er[s * D_];
        xsh[tid] = ctx;
        xsh[D_ + tid] = E[(size_t)tokens[b * T_ + t] * D_ + tid];
    }
    __syncthreads();

    // z = x@Wx + h@Wh + b   (gate order i,f,g,o; thread tid owns channel tid)
    float zi = b_lstm[tid],        zf = b_lstm[D_ + tid];
    float zg = b_lstm[2 * D_ + tid], zo = b_lstm[3 * D_ + tid];
    for (int k = 0; k < 2 * D_; ++k) {
        const float x = xsh[k];
        const float* w = Wx + (size_t)k * (4 * D_) + tid;
        zi += x * w[0]; zf += x * w[D_]; zg += x * w[2 * D_]; zo += x * w[3 * D_];
    }
    for (int k = 0; k < D_; ++k) {
        const float hv = hsh[k];
        const float* w = Wh + (size_t)k * (4 * D_) + tid;
        zi += hv * w[0]; zf += hv * w[D_]; zg += hv * w[2 * D_]; zo += hv * w[3 * D_];
    }
    const float cn = sigm(zf) * c[b * D_ + tid] + sigm(zi) * tanhf(zg);
    const float hn = sigm(zo) * tanhf(cn);
    c[b * D_ + tid] = cn;
    h[b * D_ + tid] = hn;
    hbf[b * D_ + tid] = f2bf(hn);
}

// ---------------------------------------------------------------------------
// Per step: logits = hbf @ Wo + bo via v_wmma_f32_16x16x32_bf16.
// A = hbf [32][256] bf16 (row-major), B = WoT [32000][256] bf16 (= Wo^T).
// One wave per 16x16 output tile; 8 WMMAs chain over K=256.
// grid = 500 blocks x 256 threads (8 waves) -> 4000 tiles exactly.
// ---------------------------------------------------------------------------
__global__ void __launch_bounds__(256)
k_logits_gemm(const __bf16* __restrict__ hbf, const __bf16* __restrict__ WoT,
              const float* __restrict__ bo, float* __restrict__ logits) {
    const int lane = threadIdx.x & 31;
    const int wave = threadIdx.x >> 5;
    const int gw   = blockIdx.x * 8 + wave;   // 0..3999
    const int ntile = gw >> 1;                // 0..1999
    const int mtile = gw & 1;                 // 0..1
    const bool lo  = lane < 16;
    const int  l15 = lane & 15;

    // A fragment source: row m of hbf; contiguous K order matching B:
    // lane holds K = kb + 0..15 (lanes 0-15) / kb + 16..31 (16-31)
    const __bf16* arow = hbf + (mtile * 16 + l15) * D_;
    // B fragment source: WoT row n (contiguous along K); lanes 0-15: K=kb+0..15,
    // lanes 16-31: K=kb+16..31, N = lane&15.
    const __bf16* brow = WoT + (size_t)(ntile * 16 + l15) * D_;

    f32x8 acc = {};
    #pragma unroll
    for (int kk = 0; kk < 8; ++kk) {
        const int kb = kk * 32;
        bf16x16 a, bfrag;
        const int a0 = kb + (lo ? 0 : 16);
        const int a1 = a0 + 8;
        #pragma unroll
        for (int i = 0; i < 8; ++i) { a[i] = arow[a0 + i]; a[8 + i] = arow[a1 + i]; }
        const int b0 = kb + (lo ? 0 : 16);
        #pragma unroll
        for (int i = 0; i < 16; ++i) bfrag[i] = brow[b0 + i];
        acc = __builtin_amdgcn_wmma_f32_16x16x32_bf16(
            /*neg_a=*/false, a, /*neg_b=*/false, bfrag,
            /*c_mod=*/(short)0, acc, /*reuse_a=*/false, /*reuse_b=*/false);
    }

    const int n = ntile * 16 + l15;
    const float bias = bo[n];
    const int mbase = mtile * 16 + (lo ? 0 : 8);
    #pragma unroll
    for (int r = 0; r < 8; ++r)
        logits[(size_t)(mbase + r) * V_ + n] = acc[r] + bias;
}

// ---------------------------------------------------------------------------
// Per step: row softmax over V=32000, write probs into out[b][t][:].
// One block per batch row. logits buffer is scratch (overwritten with exp).
// ---------------------------------------------------------------------------
__global__ void __launch_bounds__(256)
k_softmax_v(float* __restrict__ logits, float* __restrict__ out, int t) {
    const int b = blockIdx.x;
    const int tid = threadIdx.x;
    __shared__ float red[256];
    float* lr = logits + (size_t)b * V_;

    float mx = -3.0e38f;
    for (int v = tid; v < V_; v += 256) mx = fmaxf(mx, lr[v]);
    red[tid] = mx; __syncthreads();
    for (int s = 128; s > 0; s >>= 1) { if (tid < s) red[tid] = fmaxf(red[tid], red[tid + s]); __syncthreads(); }
    mx = red[0]; __syncthreads();

    float sum = 0.0f;
    for (int v = tid; v < V_; v += 256) { float e = expf(lr[v] - mx); lr[v] = e; sum += e; }
    red[tid] = sum; __syncthreads();
    for (int s = 128; s > 0; s >>= 1) { if (tid < s) red[tid] += red[tid + s]; __syncthreads(); }
    const float inv = 1.0f / red[0];

    float* orow = out + ((size_t)b * T_ + t) * V_;
    for (int v = tid; v < V_; v += 256) orow[v] = lr[v] * inv;
}

// ---------------------------------------------------------------------------
// Launch: 3 setup kernels + 64 x (attn_lstm, wmma gemm, softmax)
// ---------------------------------------------------------------------------
extern "C" void kernel_launch(void* const* d_in, const int* in_sizes, int n_in,
                              void* d_out, int out_size, void* d_ws, size_t ws_size,
                              hipStream_t stream) {
    const int*   tokens  = (const int*)  d_in[0];
    const float* h0      = (const float*)d_in[1];
    const float* c0      = (const float*)d_in[2];
    const float* enc_out = (const float*)d_in[3];
    const float* E       = (const float*)d_in[4];
    const float* Wq      = (const float*)d_in[5];
    const float* bq      = (const float*)d_in[6];
    const float* Wk      = (const float*)d_in[7];
    const float* bk      = (const float*)d_in[8];
    const float* Vw      = (const float*)d_in[9];
    const float* bv      = (const float*)d_in[10];
    const float* Wx      = (const float*)d_in[11];
    const float* Wh      = (const float*)d_in[12];
    const float* b_lstm  = (const float*)d_in[13];
    const float* Wo      = (const float*)d_in[14];
    const float* bo      = (const float*)d_in[15];
    float* out = (float*)d_out;

    // Workspace layout (bytes)
    char* ws = (char*)d_ws;
    __bf16* WoT    = (__bf16*)(ws);                               // 16,384,000 B
    float*  keys   = (float*) (ws + 16384000);                    //  4,194,304 B
    float*  logits = (float*) (ws + 16384000 + 4194304);          //  4,096,000 B
    float*  h      = (float*) (ws + 16384000 + 4194304 + 4096000);            // 32768 B
    float*  c      = (float*) (ws + 16384000 + 4194304 + 4096000 + 32768);    // 32768 B
    __bf16* hbf    = (__bf16*)(ws + 16384000 + 4194304 + 4096000 + 65536);    // 16384 B

    // One-time setup (per call; deterministic)
    k_transpose_wo<<<dim3(V_ / 32, D_ / 32), 256, 0, stream>>>(Wo, WoT);
    k_keys<<<B_ * S_, 256, 0, stream>>>(enc_out, Wk, bk, keys);
    k_init_state<<<B_, 256, 0, stream>>>(h0, c0, h, c);

    for (int t = 0; t < T_; ++t) {
        k_attn_lstm<<<B_, 256, 0, stream>>>(tokens, t, h, c, keys, enc_out, E,
                                            Wq, bq, Vw, bv, Wx, Wh, b_lstm, hbf);
        k_logits_gemm<<<(2 * (V_ / 16)) / 8, 256, 0, stream>>>(hbf, WoT, bo, logits);
        k_softmax_v<<<B_, 256, 0, stream>>>(logits, out, t);
    }
}
